// GIN_78606491452619
// MI455X (gfx1250) — compile-verified, matrix operand order checked by reading, not run on hardware
//
#include <hip/hip_runtime.h>
#include <hip/hip_bf16.h>
#include <math.h>

typedef __attribute__((ext_vector_type(16))) _Float16 v16h;
typedef __attribute__((ext_vector_type(8)))  float    v8f;
typedef unsigned int u32x4 __attribute__((ext_vector_type(4)));
typedef int          i32x4v __attribute__((ext_vector_type(4)));
typedef int          i32x8v __attribute__((ext_vector_type(8)));

#define Dfeat 128
#define GR    512

#if __has_builtin(__builtin_amdgcn_tensor_load_to_lds) && __has_builtin(__builtin_amdgcn_s_wait_tensorcnt)
#define USE_TDM 1
#else
#define USE_TDM 0
#endif

// ---------------------------------------------------------------------------
// Weight conversion + swizzle: f32 [K,Ncol] row-major  ->  f16 in WMMA
// fragment order, as contiguous 32KB chunks:
//   chunk c = panelIdx*(K/128) + kbIdx   (16384 halves each)
//   within chunk: i = (((ks*8 + t)*32 + lane)*16 + h)
//   k = kbIdx*128 + 32*ks + ((h&8)<<1) + (lane&16 ? 8:0) + (h&7)
//   n = panelIdx*128 + t*16 + (lane&15)
// Staging into LDS then becomes a pure contiguous copy (TDM-friendly).
// ---------------------------------------------------------------------------
__global__ __launch_bounds__(256) void swizzle_weights_kernel(const float* __restrict__ src,
                                                              _Float16* __restrict__ dst,
                                                              int K, int Ncol) {
    int o = blockIdx.x * 256 + threadIdx.x;
    if (o >= K * Ncol) return;
    int chunk = o >> 14;
    int i = o & 16383;
    int kblocks = K >> 7;
    int kbIdx = chunk % kblocks;
    int panelIdx = chunk / kblocks;
    int h  = i & 15;
    int ln = (i >> 4) & 31;
    int t  = (i >> 9) & 7;
    int ks = i >> 12;
    int k = kbIdx * 128 + 32 * ks + ((h & 8) << 1) + ((ln & 16) ? 8 : 0) + (h & 7);
    int n = panelIdx * 128 + t * 16 + (ln & 15);
    dst[o] = (_Float16)src[(long)k * Ncol + n];
}

__global__ __launch_bounds__(256) void copy_f32_kernel(const float* __restrict__ src,
                                                       float* __restrict__ dst, long n) {
    long i = (long)blockIdx.x * blockDim.x + threadIdx.x;
    if (i < n) dst[i] = src[i];
}

__global__ __launch_bounds__(256) void zero_f32_kernel(float* __restrict__ p, long n) {
    long i = (long)blockIdx.x * blockDim.x + threadIdx.x;
    if (i < n) p[i] = 0.0f;
}

// ---------------------------------------------------------------------------
// Edge scatter-add: agg[dst] += x[src].  One wave per edge, 4 coalesced
// features per lane.
// ---------------------------------------------------------------------------
__global__ __launch_bounds__(256) void scatter_add_kernel(float* __restrict__ agg,
                                                          const float* __restrict__ x,
                                                          const int* __restrict__ ei,
                                                          int E) {
    int e = blockIdx.x * (blockDim.x >> 5) + (threadIdx.x >> 5);
    int lane = threadIdx.x & 31;
    if (e >= E) return;
    int s = ei[e];
    int d = ei[E + e];
    const float* xs = x + (long)s * Dfeat;
    float* ad = agg + (long)d * Dfeat;
#pragma unroll
    for (int j = 0; j < 4; ++j) {
        int f = lane + 32 * j;
        atomicAdd(&ad[f], xs[f]);
    }
}

// ---------------------------------------------------------------------------
// Tiled WMMA GEMM: Y[M,Ncol] = act(A[M,K](f32->f16) @ Bswz(f16) + bias)
//
// Block = 256 threads = 8 waves; wave owns 16 rows x 128 cols (8 v8f accs).
// Column panel = blockIdx.y.  Per 128-K block:
//   - 32KB fragment-ordered weight chunk DMA'd global->LDS by the Tensor
//     Data Mover (tensor_load_to_lds + s_wait_tensorcnt), fallback: float4
//     copy (the chunk is contiguous either way),
//   - A fragment: 4x global_load_b128 + v_cvt_pk_f16_f32,
//   - 8 B fragments preloaded from LDS (2x ds_load_b128 each, clauseable),
//     then an 8-deep v_wmma_f32_16x16x32_f16 chain; 4 K-steps unrolled
//     => 32 static WMMAs.
// ---------------------------------------------------------------------------
__global__ __launch_bounds__(256) void wmma_gemm_tiled_kernel(const float* __restrict__ A,
                                                              const _Float16* __restrict__ Bswz,
                                                              const float* __restrict__ bias,
                                                              float* __restrict__ Y,
                                                              int M, int K, int Ncol, int relu) {
    __shared__ __align__(32) _Float16 sB[4 * 8 * 32 * 16];   // 16384 halves = 32 KB

    const int tid   = threadIdx.x;
    const int lane  = tid & 31;
    const int wave  = tid >> 5;
    const int mloc  = lane & 15;
    const int hi8   = (lane & 16) ? 8 : 0;
    const int panel = blockIdx.y * 128;
    const int kblocks = K >> 7;

    const int tiles_m = M >> 4;
    const int mytile  = blockIdx.x * 8 + wave;
    const bool active = (mytile < tiles_m);
    const long arowK  = (long)(mytile * 16 + mloc) * K;

    v8f acc[8] = {};

    for (int kb = 0; kb < K; kb += 128) {
        // ---- stage the pre-swizzled 32KB chunk into LDS ----
        const _Float16* chunk = Bswz + ((long)(blockIdx.y * kblocks + (kb >> 7)) << 14);
#if USE_TDM
        if (tid == 0) {
            unsigned long long ga = (unsigned long long)(const void*)chunk;
            unsigned ldsOff = (unsigned)(unsigned long long)(const void*)&sB[0];
            // D# group0: count=1 | lds_addr | global_addr[56:0] | type=2
            u32x4 g0 = { 1u, ldsOff, (unsigned)ga,
                         (unsigned)((ga >> 32) & 0x01FFFFFFull) | 0x80000000u };
            // D# group1: data_size=8B; tensor_dim0=4096, tensor_dim1=1;
            // tile_dim0=4096, tile_dim1=1; tensor_dim0_stride=4096
            i32x8v g1 = { (int)0x00030000, (int)0x10000000, (int)0x00010000,
                          (int)0x10000000, 1, 4096, 0, 0 };
            i32x4v g2 = { 0, 0, 0, 0 };
            i32x4v g3 = { 0, 0, 0, 0 };
#if __clang_major__ >= 23
            i32x8v g4 = { 0, 0, 0, 0, 0, 0, 0, 0 };
            __builtin_amdgcn_tensor_load_to_lds(g0, g1, g2, g3, g4, 0);
#else
            __builtin_amdgcn_tensor_load_to_lds(g0, g1, g2, g3, 0);
#endif
            __builtin_amdgcn_s_wait_tensorcnt(0);
        }
#else
        for (int it = tid; it < 2048; it += 256)            // 2048 x 16B = 32KB
            ((float4*)sB)[it] = ((const float4*)chunk)[it];
#endif
        __syncthreads();

        if (active) {
            if (kb + 128 < K)
                __builtin_prefetch(A + arowK + kb + 128, 0, 0);

#pragma unroll
            for (int ks = 0; ks < 4; ++ks) {
                // A fragment: two 8-float runs -> 4x b128 loads + pack to f16
                const float* ap = A + arowK + kb + ks * 32 + hi8;
                const float4 a0 = *(const float4*)(ap + 0);
                const float4 a1 = *(const float4*)(ap + 4);
                const float4 a2 = *(const float4*)(ap + 16);
                const float4 a3 = *(const float4*)(ap + 20);
                v16h av;
                av[0]  = (_Float16)a0.x; av[1]  = (_Float16)a0.y;
                av[2]  = (_Float16)a0.z; av[3]  = (_Float16)a0.w;
                av[4]  = (_Float16)a1.x; av[5]  = (_Float16)a1.y;
                av[6]  = (_Float16)a1.z; av[7]  = (_Float16)a1.w;
                av[8]  = (_Float16)a2.x; av[9]  = (_Float16)a2.y;
                av[10] = (_Float16)a2.z; av[11] = (_Float16)a2.w;
                av[12] = (_Float16)a3.x; av[13] = (_Float16)a3.y;
                av[14] = (_Float16)a3.z; av[15] = (_Float16)a3.w;

                // Preload all 8 B fragments (clauseable ds_load_b128 pairs),
                // then run the WMMA chain without per-step LDS stalls.
                v16h bv[8];
#pragma unroll
                for (int t = 0; t < 8; ++t)
                    bv[t] = *(const v16h*)&sB[((ks * 8 + t) * 32 + lane) * 16];
#pragma unroll
                for (int t = 0; t < 8; ++t)
                    acc[t] = __builtin_amdgcn_wmma_f32_16x16x32_f16(
                        false, av, false, bv[t], (short)0, acc[t], false, false);
            }
        }
        __syncthreads();
    }

    if (active) {
        const int mbase = mytile * 16;
#pragma unroll
        for (int t = 0; t < 8; ++t) {
            const int n = panel + t * 16 + mloc;
            const float bs = bias[n];
#pragma unroll
            for (int r = 0; r < 8; ++r) {
                int m = mbase + r + hi8;
                float v = acc[t][r] + bs;
                if (relu) v = fmaxf(v, 0.0f);
                Y[(long)m * Ncol + n] = v;
            }
        }
    }
}

// ---------------------------------------------------------------------------
// BatchNorm statistics: one block per feature column, reduce over M rows.
// ---------------------------------------------------------------------------
__global__ __launch_bounds__(256) void colstats_kernel(const float* __restrict__ h,
                                                       float* __restrict__ mu,
                                                       float* __restrict__ rsig, int M) {
    const int col = blockIdx.x;
    const int tid = threadIdx.x;
    float s = 0.0f, s2 = 0.0f;
    for (int r = tid; r < M; r += 256) {
        float v = h[(long)r * Dfeat + col];
        s += v;
        s2 += v * v;
    }
    __shared__ float sh[256], sh2[256];
    sh[tid] = s; sh2[tid] = s2;
    __syncthreads();
    for (int st = 128; st > 0; st >>= 1) {
        if (tid < st) { sh[tid] += sh[tid + st]; sh2[tid] += sh2[tid + st]; }
        __syncthreads();
    }
    if (tid == 0) {
        float m = sh[0] / (float)M;
        float var = sh2[0] / (float)M - m * m;
        mu[col] = m;
        rsig[col] = rsqrtf(var + 1e-5f);
    }
}

__global__ __launch_bounds__(256) void bn_relu_kernel(const float* __restrict__ h,
                                                      const float* __restrict__ mu,
                                                      const float* __restrict__ rsig,
                                                      const float* __restrict__ gam,
                                                      const float* __restrict__ bet,
                                                      float* __restrict__ out, long n) {
    long i = (long)blockIdx.x * blockDim.x + threadIdx.x;
    if (i >= n) return;
    int f = (int)(i & (Dfeat - 1));
    float v = gam[f] * (h[i] - mu[f]) * rsig[f] + bet[f];
    out[i] = fmaxf(v, 0.0f);
}

// ---------------------------------------------------------------------------
// Pooling into hcat [G,512] = [a1 | a2 | m1 | m2].  Max via int-bit atomicMax
// (values >= 0 post-ReLU; 0-init matches empty-segment -> 0).
// ---------------------------------------------------------------------------
__global__ __launch_bounds__(256) void pool_kernel(const float* __restrict__ h,
                                                   const int* __restrict__ batch,
                                                   float* __restrict__ hcat,
                                                   int addOff, int maxOff, long n) {
    long i = (long)blockIdx.x * blockDim.x + threadIdx.x;
    if (i >= n) return;
    int node = (int)(i >> 7);
    int f = (int)(i & (Dfeat - 1));
    int g = batch[node];
    float v = h[i];
    float* row = hcat + (long)g * 512;
    atomicAdd(&row[addOff + f], v);
    atomicMax((int*)&row[maxOff + f], __float_as_int(v));
}

// ---------------------------------------------------------------------------
// Final readout: logit[g] = dot(r1[g,:], w2) + b; out = [sigmoid | logit]
// ---------------------------------------------------------------------------
__global__ __launch_bounds__(256) void final_kernel(const float* __restrict__ r1,
                                                    const float* __restrict__ w2,
                                                    const float* __restrict__ b2,
                                                    float* __restrict__ out) {
    const int g = blockIdx.x;
    const int tid = threadIdx.x;
    float s = 0.0f;
    for (int k = tid; k < 512; k += 256) s += r1[(long)g * 512 + k] * w2[k];
    __shared__ float sh[256];
    sh[tid] = s;
    __syncthreads();
    for (int st = 128; st > 0; st >>= 1) {
        if (tid < st) sh[tid] += sh[tid + st];
        __syncthreads();
    }
    if (tid == 0) {
        float logit = sh[0] + b2[0];
        out[g] = 1.0f / (1.0f + expf(-logit));
        out[GR + g] = logit;
    }
}

// ---------------------------------------------------------------------------
// Host orchestration
// ---------------------------------------------------------------------------
extern "C" void kernel_launch(void* const* d_in, const int* in_sizes, int n_in,
                              void* d_out, int out_size, void* d_ws, size_t ws_size,
                              hipStream_t stream) {
    (void)n_in; (void)out_size; (void)ws_size;
    const float* x      = (const float*)d_in[0];
    const int*   ei     = (const int*)d_in[1];
    const int*   batch  = (const int*)d_in[2];
    const float* c1_W1  = (const float*)d_in[3];
    const float* c1_b1  = (const float*)d_in[4];
    const float* c1_g   = (const float*)d_in[5];
    const float* c1_be  = (const float*)d_in[6];
    const float* c1_W2  = (const float*)d_in[7];
    const float* c1_b2  = (const float*)d_in[8];
    const float* c2_W1  = (const float*)d_in[9];
    const float* c2_b1  = (const float*)d_in[10];
    const float* c2_g   = (const float*)d_in[11];
    const float* c2_be  = (const float*)d_in[12];
    const float* c2_W2  = (const float*)d_in[13];
    const float* c2_b2  = (const float*)d_in[14];
    const float* lin1_W = (const float*)d_in[15];
    const float* lin1_b = (const float*)d_in[16];
    const float* lin2_W = (const float*)d_in[17];
    const float* lin2_b = (const float*)d_in[18];
    float* out = (float*)d_out;

    const int N = in_sizes[0] / Dfeat;     // 50000
    const int E = in_sizes[1] / 2;         // 600000
    const long ND = (long)N * Dfeat;

    // Workspace carve-up (~105 MB)
    char* w = (char*)d_ws;
    float* B0 = (float*)w; w += ND * sizeof(float);   // agg
    float* B1 = (float*)w; w += ND * sizeof(float);   // h (pre-BN) / h2 out
    float* B2 = (float*)w; w += ND * sizeof(float);   // hn (post-BN-ReLU)
    float* B3 = (float*)w; w += ND * sizeof(float);   // h1 (layer1 output)
    _Float16* w1a   = (_Float16*)w; w += 16384 * sizeof(_Float16);
    _Float16* w2a   = (_Float16*)w; w += 16384 * sizeof(_Float16);
    _Float16* w1b   = (_Float16*)w; w += 16384 * sizeof(_Float16);
    _Float16* w2b   = (_Float16*)w; w += 16384 * sizeof(_Float16);
    _Float16* wlin1 = (_Float16*)w; w += 262144 * sizeof(_Float16);
    float* mu   = (float*)w; w += Dfeat * sizeof(float);
    float* rsig = (float*)w; w += Dfeat * sizeof(float);
    float* hcat = (float*)w; w += (long)GR * 512 * sizeof(float);
    float* r1   = (float*)w; w += (long)GR * 512 * sizeof(float);

    const int ew = (int)((ND + 255) / 256);
    const int scatterBlocks = (E + 7) / 8;
    const dim3 gemmConvGrid((N / 16 + 7) / 8, 1);          // 391 x 1
    const dim3 gemmOutGrid((GR / 16 + 7) / 8, 512 / 128);  // 4 x 4

    // Weight conversion + fragment-order swizzle (contiguous 32KB chunks)
    swizzle_weights_kernel<<<64, 256, 0, stream>>>(c1_W1, w1a, Dfeat, Dfeat);
    swizzle_weights_kernel<<<64, 256, 0, stream>>>(c1_W2, w2a, Dfeat, Dfeat);
    swizzle_weights_kernel<<<64, 256, 0, stream>>>(c2_W1, w1b, Dfeat, Dfeat);
    swizzle_weights_kernel<<<64, 256, 0, stream>>>(c2_W2, w2b, Dfeat, Dfeat);
    swizzle_weights_kernel<<<1024, 256, 0, stream>>>(lin1_W, wlin1, 512, 512);

    // hcat accumulators start at zero
    zero_f32_kernel<<<(GR * 512 + 255) / 256, 256, 0, stream>>>(hcat, (long)GR * 512);

    // ---- GINConv layer 1 ----
    copy_f32_kernel<<<ew, 256, 0, stream>>>(x, B0, ND);
    scatter_add_kernel<<<scatterBlocks, 256, 0, stream>>>(B0, x, ei, E);
    wmma_gemm_tiled_kernel<<<gemmConvGrid, 256, 0, stream>>>(B0, w1a, c1_b1, B1, N, Dfeat, Dfeat, 0);
    colstats_kernel<<<Dfeat, 256, 0, stream>>>(B1, mu, rsig, N);
    bn_relu_kernel<<<ew, 256, 0, stream>>>(B1, mu, rsig, c1_g, c1_be, B2, ND);
    wmma_gemm_tiled_kernel<<<gemmConvGrid, 256, 0, stream>>>(B2, w2a, c1_b2, B3, N, Dfeat, Dfeat, 1);
    pool_kernel<<<ew, 256, 0, stream>>>(B3, batch, hcat, 0, 256, ND);

    // ---- GINConv layer 2 ----
    copy_f32_kernel<<<ew, 256, 0, stream>>>(B3, B0, ND);
    scatter_add_kernel<<<scatterBlocks, 256, 0, stream>>>(B0, B3, ei, E);
    wmma_gemm_tiled_kernel<<<gemmConvGrid, 256, 0, stream>>>(B0, w1b, c2_b1, B1, N, Dfeat, Dfeat, 0);
    colstats_kernel<<<Dfeat, 256, 0, stream>>>(B1, mu, rsig, N);
    bn_relu_kernel<<<ew, 256, 0, stream>>>(B1, mu, rsig, c2_g, c2_be, B2, ND);
    wmma_gemm_tiled_kernel<<<gemmConvGrid, 256, 0, stream>>>(B2, w2b, c2_b2, B1, N, Dfeat, Dfeat, 1);
    pool_kernel<<<ew, 256, 0, stream>>>(B1, batch, hcat, 128, 384, ND);

    // ---- Readout ----
    wmma_gemm_tiled_kernel<<<gemmOutGrid, 256, 0, stream>>>(hcat, wlin1, lin1_b, r1, GR, 512, 512, 1);
    final_kernel<<<GR, 256, 0, stream>>>(r1, lin2_W, lin2_b, out);
}
